// MSCSA_35218731827434
// MI455X (gfx1250) — 8-bit matrix operand order repaired and checked by reading, compiled, not run on hardware
//
#include <hip/hip_runtime.h>
#include <hip/hip_bf16.h>
#include <math.h>
#include <stdint.h>

typedef __attribute__((ext_vector_type(8)))  int   v8i;
typedef __attribute__((ext_vector_type(16))) int   v16i;
typedef __attribute__((ext_vector_type(8)))  float v8f;

#define BATCH 32
#define CH    64
#define HT    128
#define WD    128
#define ROWS  2
#define SROWS 6
#define SCOLS 132
#define BRSZ  (SROWS*SCOLS*CH)
#define WBYTES (35*64*64)
#define LDSW_OFF (3*BRSZ)
#define MAIN_LDS (3*BRSZ + WBYTES)
#define THREADS 512

#define FP8_POS 0x38u
#define FP8_NEG 0xB8u
#define FP8_ZERO 0x00u

static constexpr size_t S_BYTES    = (size_t)BATCH*CH*HT*WD*4;
static constexpr size_t GAP_OFF    = S_BYTES;
static constexpr size_t GATE_OFF   = GAP_OFF  + 2048*4;
static constexpr size_t SCALEA_OFF = GATE_OFF + 2048*4;
static constexpr size_t BIASB_OFF  = SCALEA_OFF + 3*64*4;
static constexpr size_t WS1_OFF    = BIASB_OFF  + 3*64*4;
static constexpr size_t WS3_OFF    = WS1_OFF + (size_t)1*64*64;
static constexpr size_t WS5_OFF    = WS3_OFF + (size_t)9*64*64;

__global__ void zero_gap(float* __restrict__ gap) {
    int i = blockIdx.x * 256 + threadIdx.x;
    if (i < 2048) gap[i] = 0.f;
}

__global__ __launch_bounds__(256)
void prep_weights(const float* __restrict__ w, const float* __restrict__ cb,
                  const float* __restrict__ gamma, const float* __restrict__ beta,
                  const float* __restrict__ rm, const float* __restrict__ rv,
                  int KS, unsigned char* __restrict__ wsign,
                  float* __restrict__ scaleA_b, float* __restrict__ biasB_b) {
    __shared__ float red_a[256];
    __shared__ float red_w[256];
    const int co  = blockIdx.x;
    const int tid = threadIdx.x;
    const int kk  = KS * KS;
    const int cnt = CH * kk;

    float sa = 0.f, sw = 0.f;
    for (int e = tid; e < cnt; e += 256) {
        float v = w[co * cnt + e];
        sa += fabsf(v);
        sw += v;
    }
    red_a[tid] = sa; red_w[tid] = sw;
    __syncthreads();
    for (int off = 128; off > 0; off >>= 1) {
        if (tid < off) { red_a[tid] += red_a[tid + off]; red_w[tid] += red_w[tid + off]; }
        __syncthreads();
    }
    const float mabs = red_a[0] / (float)cnt;
    const float mw   = red_w[0] / (float)cnt;

    for (int e = tid; e < cnt; e += 256) {
        const int ci = e / kk;
        const int t  = e % kk;
        float d = w[co * cnt + e] - mw;
        unsigned char sg = (d > 0.f) ? FP8_POS : ((d < 0.f) ? FP8_NEG : FP8_ZERO);
        wsign[(size_t)t * 4096 + co * 64 + ci] = sg;
    }
    if (tid == 0) {
        float inv = gamma[co] * rsqrtf(rv[co] + 1e-5f);
        scaleA_b[co] = mabs * inv;
        biasB_b[co]  = cb[co] * inv + beta[co] - rm[co] * inv;
    }
}

#define LOAD_A64(dst, off0, a32) {                     \
    int2 _d0 = *(const int2*)(a32);                    \
    int2 _d1 = *(const int2*)((a32) + 2);              \
    int2 _d2 = *(const int2*)((a32) + 8);              \
    int2 _d3 = *(const int2*)((a32) + 10);             \
    dst[(off0)+0] = _d0.x; dst[(off0)+1] = _d0.y;      \
    dst[(off0)+2] = _d1.x; dst[(off0)+3] = _d1.y;      \
    dst[(off0)+4] = _d2.x; dst[(off0)+5] = _d2.y;      \
    dst[(off0)+6] = _d3.x; dst[(off0)+7] = _d3.y; }

#define LOAD_B64(dst, off0, b32) {                                         \
    int4 _q0 = *(const int4*)(b32);                                        \
    int4 _q1 = *(const int4*)((b32) + 8);                                  \
    dst[(off0)+0] = _q0.x; dst[(off0)+1] = _q0.y;                          \
    dst[(off0)+2] = _q0.z; dst[(off0)+3] = _q0.w;                          \
    dst[(off0)+4] = _q1.x; dst[(off0)+5] = _q1.y;                          \
    dst[(off0)+6] = _q1.z; dst[(off0)+7] = _q1.w; }

template<int KS>
__device__ __forceinline__
void branch_accum(const char* __restrict__ lds_buf,
                  const char* __restrict__ lws,
                  const float* __restrict__ scaleA_b,
                  float fa[4][8], int ro, int wo_base, int mlane, int khalf) {
    constexpr int PAD = KS / 2;
    constexpr int NTAP = KS * KS;
    constexpr int NPAIR = NTAP / 2;
    const v8f vzero = {0.f,0.f,0.f,0.f,0.f,0.f,0.f,0.f};
    v8f acc[4];
#pragma unroll
    for (int t = 0; t < 4; ++t) acc[t] = vzero;

    const int cbase = wo_base + mlane - PAD + 2;
    const int rbase = ro - PAD + 2;

#pragma unroll
    for (int p = 0; p < NPAIR; ++p) {
        const int t0 = 2 * p, t1 = 2 * p + 1;
        const int r0 = rbase + t0 / KS, c0 = cbase + t0 % KS;
        const int r1 = rbase + t1 / KS, c1 = cbase + t1 % KS;
        const int* a0 = (const int*)(lds_buf + ((r0 * SCOLS + c0) * 64 + khalf * 16));
        const int* a1 = (const int*)(lds_buf + ((r1 * SCOLS + c1) * 64 + khalf * 16));
        v16i a;
        LOAD_A64(a, 0, a0)
        LOAD_A64(a, 8, a1)
#pragma unroll
        for (int nt = 0; nt < 4; ++nt) {
            const int co = nt * 16 + mlane;
            const int* b0 = (const int*)(lws + t0 * 4096 + co * 64 + khalf * 16);
            const int* b1 = b0 + 1024;
            v16i bf;
            LOAD_B64(bf, 0, b0)
            LOAD_B64(bf, 8, b1)
            acc[nt] = __builtin_amdgcn_wmma_f32_16x16x128_fp8_fp8(
                a, bf, (short)0, acc[nt], false, false);
        }
    }
    if (NTAP & 1) {
        const int t = NTAP - 1;
        const int r = rbase + t / KS, c = cbase + t % KS;
        const int* a32 = (const int*)(lds_buf + ((r * SCOLS + c) * 64 + khalf * 16));
        v8i a;
        LOAD_A64(a, 0, a32)
#pragma unroll
        for (int nt = 0; nt < 4; ++nt) {
            const int co = nt * 16 + mlane;
            const int* b32 = (const int*)(lws + t * 4096 + co * 64 + khalf * 16);
            v8i bf;
            LOAD_B64(bf, 0, b32)
            acc[nt] = __builtin_amdgcn_wmma_f32_16x16x64_fp8_fp8(
                a, bf, (short)0, acc[nt], false, false);
        }
    }
#pragma unroll
    for (int nt = 0; nt < 4; ++nt) {
        const float sA = scaleA_b[nt * 16 + mlane];
#pragma unroll
        for (int e = 0; e < 8; ++e) fa[nt][e] += acc[nt][e] * sA;
    }
}

__global__ __launch_bounds__(THREADS, 1)
void bconv_main(const float* __restrict__ x,
                const float* __restrict__ move1, const float* __restrict__ move3,
                const float* __restrict__ move5,
                const char* __restrict__ wsign_all,
                const float* __restrict__ scaleA, const float* __restrict__ biasB,
                float* __restrict__ s_out, float* __restrict__ gap) {
    extern __shared__ char smem[];
    char* buf0 = smem;
    char* buf1 = smem + BRSZ;
    char* buf2 = smem + 2 * BRSZ;
    char* lweights = smem + LDSW_OFF;
    __shared__ float gap_lds[64];

    const int tid = threadIdx.x;
    const int bid = blockIdx.x;
    const int n  = bid >> 6;
    const int h0 = (bid & 63) * ROWS;

    if (tid < 64) gap_lds[tid] = 0.f;

    {
        const unsigned lbase = (unsigned)(uintptr_t)lweights;
        for (int idx = tid; idx < WBYTES / 16; idx += THREADS) {
            const unsigned goff  = (unsigned)idx * 16u;
            const unsigned laddr = lbase + goff;
            asm volatile("global_load_async_to_lds_b128 %0, %1, %2"
                         :: "v"(laddr), "v"(goff), "s"(wsign_all)
                         : "memory");
        }
    }

    for (int idx = tid; idx < SROWS * SCOLS * 16; idx += THREADS) {
        const int q   = idx / (SROWS * SCOLS);
        const int pos = idx % (SROWS * SCOLS);
        const int ci4 = q * 4;
        const int r = pos / SCOLS;
        const int c = pos % SCOLS;
        const int hh = h0 + r - 2;
        const int ww = c - 2;
        const bool inb = (hh >= 0) & (hh < HT) & (ww >= 0) & (ww < WD);
        unsigned int p0 = 0, p1 = 0, p2 = 0;
#pragma unroll
        for (int j = 0; j < 4; ++j) {
            const int ci = ci4 + j;
            float xv = 0.f;
            if (inb) xv = x[(((size_t)n * CH + ci) * HT + hh) * WD + ww];
            float v1 = xv + move1[ci];
            float v3 = xv + move3[ci];
            float v5 = xv + move5[ci];
            unsigned s1 = inb ? ((v1 > 0.f) ? FP8_POS : ((v1 < 0.f) ? FP8_NEG : FP8_ZERO)) : FP8_ZERO;
            unsigned s3 = inb ? ((v3 > 0.f) ? FP8_POS : ((v3 < 0.f) ? FP8_NEG : FP8_ZERO)) : FP8_ZERO;
            unsigned s5 = inb ? ((v5 > 0.f) ? FP8_POS : ((v5 < 0.f) ? FP8_NEG : FP8_ZERO)) : FP8_ZERO;
            p0 |= s1 << (8 * j);
            p1 |= s3 << (8 * j);
            p2 |= s5 << (8 * j);
        }
        const int di = pos * 16 + q;
        ((unsigned int*)buf0)[di] = p0;
        ((unsigned int*)buf1)[di] = p1;
        ((unsigned int*)buf2)[di] = p2;
    }

    asm volatile("s_wait_asynccnt 0x0" ::: "memory");
    __syncthreads();

    const int wave  = tid >> 5;
    const int lane  = tid & 31;
    const int mlane = lane & 15;
    const int khalf = lane >> 4;
    const int ro      = wave >> 3;
    const int wo_base = (wave & 7) << 4;

    float fa[4][8];
#pragma unroll
    for (int t = 0; t < 4; ++t)
#pragma unroll
        for (int e = 0; e < 8; ++e) fa[t][e] = 0.f;

    branch_accum<1>(buf0, lweights,             scaleA +   0, fa, ro, wo_base, mlane, khalf);
    branch_accum<3>(buf1, lweights + 1 * 4096,  scaleA +  64, fa, ro, wo_base, mlane, khalf);
    branch_accum<5>(buf2, lweights + 10 * 4096, scaleA + 128, fa, ro, wo_base, mlane, khalf);

    const int h = h0 + ro;
#pragma unroll
    for (int nt = 0; nt < 4; ++nt) {
        const int co = nt * 16 + mlane;
        const float bias = biasB[co] + biasB[64 + co] + biasB[128 + co];
        float part = 0.f;
#pragma unroll
        for (int e = 0; e < 8; ++e) {
            const int m = e + khalf * 8;
            const int w = wo_base + m;
            const float sv = fa[nt][e] + bias;
            s_out[(((size_t)n * HT + h) * WD + w) * CH + co] = sv;
            part += sv;
        }
        atomicAdd(&gap_lds[co], part);
    }
    __syncthreads();
    if (tid < 64) atomicAdd(&gap[n * 64 + tid], gap_lds[tid]);
}

__global__ __launch_bounds__(256)
void gate_kernel(const float* __restrict__ gap, const float* __restrict__ gw,
                 const float* __restrict__ gb, float* __restrict__ gate) {
    const float inv_hw = 1.f / (float)(HT * WD);
    for (int f = blockIdx.x * 256 + threadIdx.x; f < BATCH * CH; f += gridDim.x * 256) {
        const int b  = f / CH;
        const int cp = f % CH;
        const int j = cp >> 4, i = cp & 15;
        const int c = i * 4 + j;
        const int g = c >> 4, o = c & 15;
        float dot = 0.f;
#pragma unroll
        for (int ii = 0; ii < 16; ++ii)
            dot += gap[b * CH + g * 16 + ii] * inv_hw * gw[(g * 16 + o) * 16 + ii];
        dot += gb[c];
        gate[f] = 1.f / (1.f + expf(-dot));
    }
}

__global__ __launch_bounds__(256)
void apply_gate(const float* __restrict__ x, const float* __restrict__ s,
                const float* __restrict__ gate, float* __restrict__ out) {
    const size_t t4 = (size_t)blockIdx.x * 256 + threadIdx.x;
    const size_t base = t4 * 4;
    const int n  = (int)(base >> 20);
    const int c  = (int)((base >> 14) & 63);
    const int hw = (int)(base & 16383);
    const int h = hw >> 7, w0 = hw & 127;
    const float g = gate[n * 64 + c];
    const float4 xv = *(const float4*)(x + base);
    const float* sp = s + (((size_t)n * HT + h) * WD + w0) * CH + c;
    __builtin_prefetch(sp + 4 * CH, 0, 0);
    float4 o;
    o.x = sp[0 * CH] * g + xv.x;
    o.y = sp[1 * CH] * g + xv.y;
    o.z = sp[2 * CH] * g + xv.z;
    o.w = sp[3 * CH] * g + xv.w;
    *(float4*)(out + base) = o;
}

extern "C" void kernel_launch(void* const* d_in, const int* in_sizes, int n_in,
                              void* d_out, int out_size, void* d_ws, size_t ws_size,
                              hipStream_t stream) {
    const float* x     = (const float*)d_in[0];
    const float* move1 = (const float*)d_in[1];
    const float* w1    = (const float*)d_in[2];
    const float* cb1   = (const float*)d_in[3];
    const float* ga1   = (const float*)d_in[4];
    const float* be1   = (const float*)d_in[5];
    const float* rm1   = (const float*)d_in[6];
    const float* rv1   = (const float*)d_in[7];
    const float* move3 = (const float*)d_in[8];
    const float* w3    = (const float*)d_in[9];
    const float* cb3   = (const float*)d_in[10];
    const float* ga3   = (const float*)d_in[11];
    const float* be3   = (const float*)d_in[12];
    const float* rm3   = (const float*)d_in[13];
    const float* rv3   = (const float*)d_in[14];
    const float* move5 = (const float*)d_in[15];
    const float* w5    = (const float*)d_in[16];
    const float* cb5   = (const float*)d_in[17];
    const float* ga5   = (const float*)d_in[18];
    const float* be5   = (const float*)d_in[19];
    const float* rm5   = (const float*)d_in[20];
    const float* rv5   = (const float*)d_in[21];
    const float* gw    = (const float*)d_in[22];
    const float* gb    = (const float*)d_in[23];

    char* ws = (char*)d_ws;
    float* s_buf  = (float*)(ws);
    float* gap    = (float*)(ws + GAP_OFF);
    float* gate   = (float*)(ws + GATE_OFF);
    float* scaleA = (float*)(ws + SCALEA_OFF);
    float* biasB  = (float*)(ws + BIASB_OFF);
    unsigned char* ws1 = (unsigned char*)(ws + WS1_OFF);
    unsigned char* ws3 = (unsigned char*)(ws + WS3_OFF);
    unsigned char* ws5 = (unsigned char*)(ws + WS5_OFF);

    hipFuncSetAttribute(reinterpret_cast<const void*>(bconv_main),
                        hipFuncAttributeMaxDynamicSharedMemorySize, MAIN_LDS);

    zero_gap<<<8, 256, 0, stream>>>(gap);
    prep_weights<<<64, 256, 0, stream>>>(w1, cb1, ga1, be1, rm1, rv1, 1, ws1, scaleA +   0, biasB +   0);
    prep_weights<<<64, 256, 0, stream>>>(w3, cb3, ga3, be3, rm3, rv3, 3, ws3, scaleA +  64, biasB +  64);
    prep_weights<<<64, 256, 0, stream>>>(w5, cb5, ga5, be5, rm5, rv5, 5, ws5, scaleA + 128, biasB + 128);

    bconv_main<<<BATCH * (HT / ROWS), THREADS, MAIN_LDS, stream>>>(
        x, move1, move3, move5, (const char*)ws1, scaleA, biasB, s_buf, gap);

    gate_kernel<<<8, 256, 0, stream>>>(gap, gw, gb, gate);

    const size_t nquads = (size_t)BATCH * CH * HT * WD / 4;
    apply_gate<<<(unsigned)(nquads / 256), 256, 0, stream>>>(x, s_buf, gate, (float*)d_out);
}
